// MultiheadAttention_41583873359917
// MI455X (gfx1250) — matrix operand order repaired and checked by reading, not compiled, not run on hardware
//
#include <hip/hip_runtime.h>
#include <hip/hip_bf16.h>

// ---------------- types for WMMA operands ----------------
typedef _Float16 h8   __attribute__((ext_vector_type(8)));
typedef _Float16 v16h __attribute__((ext_vector_type(16)));
typedef float    v8f  __attribute__((ext_vector_type(8)));

union PackH { v16h v; h8 h[2]; };

// A-fragment (16x32 f16, MxK): lane holds one row; chunks {0..7,16..23} for
// lanes 0-15, {8..15,24..31} for lanes 16-31. `p` = row_base + k0 + (lane&16?8:0).
static __device__ inline v16h load_a_frag(const _Float16* p) {
    PackH u;
    u.h[0] = *(const h8*)(p);
    u.h[1] = *(const h8*)(p + 16);
    return u.v;
}
// B-fragment (32x16 f16, KxN) from a row-major [N][K] (i.e. B^T) buffer:
// lane holds column n = lane&15, chunks K {0..7,16..23} (lanes 0-15) or {8..15,24..31}
// (lanes 16-31). `p` = n_row_base + k0 + (lane&16?8:0).
static __device__ inline v16h load_b_frag(const _Float16* p) {
    PackH u;
    u.h[0] = *(const h8*)(p);
    u.h[1] = *(const h8*)(p + 16);
    return u.v;
}

static __device__ inline v8f wmma_f16(v16h a, v16h b, v8f c) {
    return __builtin_amdgcn_wmma_f32_16x16x32_f16(
        /*neg_a=*/false, a, /*neg_b=*/false, b,
        /*c_mod=*/(short)0, c, /*reuse_a=*/false, /*reuse_b=*/false);
}

// ---------------- constants ----------------
#define EMB   1024
#define NH    16
#define DK    64
#define SEQ   2048
#define BAT   4
#define MTOT  (BAT * SEQ)         // 8192
#define HEADSZ ((size_t)SEQ * DK) // 131072 elems per (b,h) plane

// ---------------- conversion kernels ----------------
__global__ void cvt_f32_f16(const float* __restrict__ in, _Float16* __restrict__ out, int n) {
    int i = blockIdx.x * 256 + threadIdx.x;
    if (i < n) out[i] = (_Float16)in[i];
}

// Wt[n][k] = (f16) W[k][n]   (1024x1024)
__global__ void transpose_cvt(const float* __restrict__ in, _Float16* __restrict__ out) {
    int i = blockIdx.x * 256 + threadIdx.x;
    int nrow = i >> 10, k = i & 1023;
    out[i] = (_Float16)in[(size_t)k * EMB + nrow];
}

// ---------------- generic WMMA GEMM: C[M,1024] = A[M,1024] @ Bt^T ----------------
// grid = (N/64, M/128), block = 256 (8 waves, 4x2), wave tile = 32x32.
// mode 0: f32 row-major out (ldc=1024)
// mode 1: f16 head-split out  [(m>>11)*16 + (n>>6)][m&2047][n&63]
// mode 2: f16 head-split transposed out [(m>>11)*16 + (n>>6)][n&63][m&2047]
__global__ void gemm_wmma(const _Float16* __restrict__ A, const _Float16* __restrict__ Bt,
                          void* __restrict__ outp, int K, int mode) {
    const int tid = threadIdx.x;
    const int w = tid >> 5, lane = tid & 31;
    const int wm = w >> 1, wn = w & 1;
    const int m0 = blockIdx.y * 128 + wm * 32;
    const int n0 = blockIdx.x * 64  + wn * 32;
    const int lr = lane & 15;
    const int achunk = (lane & 16) ? 8 : 0;
    const int bchunk = (lane & 16) ? 8 : 0;
    const int rowb   = (lane & 16) ? 8 : 0;

    v8f c[2][2] = {};
    for (int k0 = 0; k0 < K; k0 += 32) {
        v16h a0 = load_a_frag(A + (size_t)(m0 + lr)      * K + k0 + achunk);
        v16h a1 = load_a_frag(A + (size_t)(m0 + 16 + lr) * K + k0 + achunk);
        v16h b0 = load_b_frag(Bt + (size_t)(n0 + lr)      * K + k0 + bchunk);
        v16h b1 = load_b_frag(Bt + (size_t)(n0 + 16 + lr) * K + k0 + bchunk);
        c[0][0] = wmma_f16(a0, b0, c[0][0]);
        c[0][1] = wmma_f16(a0, b1, c[0][1]);
        c[1][0] = wmma_f16(a1, b0, c[1][0]);
        c[1][1] = wmma_f16(a1, b1, c[1][1]);
    }

    #pragma unroll
    for (int i = 0; i < 2; i++) {
        #pragma unroll
        for (int j = 0; j < 2; j++) {
            const int nn = n0 + j * 16 + lr;
            if (mode == 0) {
                float* out = (float*)outp;
                #pragma unroll
                for (int r = 0; r < 8; r++) {
                    int mm = m0 + i * 16 + rowb + r;
                    out[(size_t)mm * 1024 + nn] = c[i][j][r];
                }
            } else if (mode == 1) {
                _Float16* out = (_Float16*)outp;
                #pragma unroll
                for (int r = 0; r < 8; r++) {
                    int mm = m0 + i * 16 + rowb + r;
                    size_t off = ((size_t)(mm >> 11) * 16 + (nn >> 6)) * HEADSZ
                               + (size_t)(mm & 2047) * DK + (nn & 63);
                    out[off] = (_Float16)c[i][j][r];
                }
            } else {
                _Float16* out = (_Float16*)outp;
                const int mmb = m0 + i * 16 + rowb;   // 8 consecutive rows -> contiguous s
                h8 pk;
                #pragma unroll
                for (int r = 0; r < 8; r++) pk[r] = (_Float16)c[i][j][r];
                size_t off = ((size_t)(mmb >> 11) * 16 + (nn >> 6)) * HEADSZ
                           + (size_t)(nn & 63) * SEQ + (mmb & 2047);
                *(h8*)(out + off) = pk;   // one b128 store
            }
        }
    }
}

// ---------------- fused attention: scores -> softmax -> attn out -> PV ----------------
// grid = (S/16, B*NH), block = 256 (8 waves). Dynamic LDS:
//   sc     : float  [16][2048]  131072 B  (scores; reused as PV partials)
//   ah     : f16    [16][2048]   65536 B  (softmaxed attn, f16)
//   red    : float  [256]         1024 B
//   rowred : float  [16]            64 B
#define SC_OFF   0
#define AH_OFF   131072
#define RED_OFF  196608
#define RR_OFF   197632
#define SMEM_BYTES 197696

__global__ void attn_fused(const _Float16* __restrict__ qb_, const _Float16* __restrict__ kb_,
                           const _Float16* __restrict__ vt_, const unsigned char* __restrict__ mask,
                           float* __restrict__ attn_out, _Float16* __restrict__ predh) {
    extern __shared__ char smem[];
    float*    sc     = (float*)(smem + SC_OFF);
    _Float16* ah     = (_Float16*)(smem + AH_OFF);
    float*    red    = (float*)(smem + RED_OFF);
    float*    rowred = (float*)(smem + RR_OFF);

    const int tid = threadIdx.x;
    const int w = tid >> 5, lane = tid & 31;
    const int lr = lane & 15;
    const int achunk = (lane & 16) ? 8 : 0;
    const int bchunk = (lane & 16) ? 8 : 0;
    const int rowb   = (lane & 16) ? 8 : 0;
    const int bh = blockIdx.y;
    const int b = bh >> 4, h = bh & 15;
    const int qb0 = blockIdx.x * 16;
    const size_t hoff = (size_t)bh * HEADSZ;

    // ---- pass 1: scores = (q @ k^T) / sqrt(DK) into LDS ----
    v16h qa0 = load_a_frag(qb_ + hoff + (size_t)(qb0 + lr) * DK + 0  + achunk);
    v16h qa1 = load_a_frag(qb_ + hoff + (size_t)(qb0 + lr) * DK + 32 + achunk);
    for (int jt = 0; jt < 16; jt++) {
        const int col0 = (jt * 8 + w) * 16;        // 128 column tiles across 8 waves
        v16h kb0 = load_b_frag(kb_ + hoff + (size_t)(col0 + lr) * DK + 0  + bchunk);
        v16h kb1 = load_b_frag(kb_ + hoff + (size_t)(col0 + lr) * DK + 32 + bchunk);
        v8f c = {};
        c = wmma_f16(qa0, kb0, c);
        c = wmma_f16(qa1, kb1, c);
        #pragma unroll
        for (int r = 0; r < 8; r++)
            sc[(rowb + r) * SEQ + col0 + lr] = c[r] * 0.125f;   // 1/sqrt(64)
    }
    __syncthreads();

    // ---- pass 2: mask + softmax, write attn (f32 -> HBM once, f16 -> LDS) ----
    const int row = tid >> 4;     // 16 threads per row
    const int seg = tid & 15;
    float* srow = sc + row * SEQ;
    const unsigned char* mrow = mask + ((size_t)b * SEQ + (qb0 + row)) * SEQ;

    float lmax = -3.0e38f;
    for (int i = 0; i < 128; i++) {
        const int cidx = i * 16 + seg;             // coalesced across the 16 threads
        float v0 = srow[cidx];
        if (mrow[cidx]) v0 = -1.0e9f;
        srow[cidx] = v0;
        lmax = fmaxf(lmax, v0);
    }
    red[tid] = lmax; __syncthreads();
    if (seg == 0) {
        float m = red[tid];
        for (int i = 1; i < 16; i++) m = fmaxf(m, red[tid + i]);
        rowred[row] = m;
    }
    __syncthreads();
    const float rmax = rowred[row];

    float lsum = 0.f;
    for (int i = 0; i < 128; i++) {
        const int cidx = i * 16 + seg;
        float e = __expf(srow[cidx] - rmax);
        srow[cidx] = e;
        lsum += e;
    }
    red[tid] = lsum; __syncthreads();
    if (seg == 0) {
        float s = 0.f;
        for (int i = 0; i < 16; i++) s += red[tid + i];
        rowred[row] = s;
    }
    __syncthreads();
    const float inv = 1.0f / rowred[row];
    float* grow = attn_out + ((size_t)bh * SEQ + (qb0 + row)) * SEQ;
    for (int i = 0; i < 128; i++) {
        const int cidx = i * 16 + seg;
        float p = srow[cidx] * inv;
        grow[cidx] = p;
        ah[row * SEQ + cidx] = (_Float16)p;
    }
    __syncthreads();

    // ---- pass 3: pred_tile[16,64] = attn[16,2048] @ v[2048,64] (v stored transposed) ----
    v8f ct[4] = {};
    const int kw = w * 256;                         // per-wave K slice
    for (int kk = 0; kk < 8; kk++) {
        const int k0 = kw + kk * 32;
        v16h a = load_a_frag(ah + lr * SEQ + k0 + achunk);          // LDS reads
        #pragma unroll
        for (int t = 0; t < 4; t++) {
            v16h bf = load_b_frag(vt_ + hoff + (size_t)(t * 16 + lr) * SEQ + k0 + bchunk);
            ct[t] = wmma_f16(a, bf, ct[t]);
        }
    }
    float* pv = sc;                                 // reuse scores buffer for partials
    #pragma unroll
    for (int t = 0; t < 4; t++)
        #pragma unroll
        for (int r = 0; r < 8; r++)
            pv[w * 1024 + (rowb + r) * 64 + t * 16 + lr] = ct[t][r];
    __syncthreads();

    for (int e = tid; e < 1024; e += 256) {
        float s = 0.f;
        #pragma unroll
        for (int ww = 0; ww < 8; ww++) s += pv[ww * 1024 + e];
        const int r = e >> 6, dc = e & 63;
        predh[((size_t)b * SEQ + qb0 + r) * (NH * DK) + h * DK + dc] = (_Float16)s;
    }
}

// ---------------- host launch ----------------
extern "C" void kernel_launch(void* const* d_in, const int* in_sizes, int n_in,
                              void* d_out, int out_size, void* d_ws, size_t ws_size,
                              hipStream_t stream) {
    (void)in_sizes; (void)n_in; (void)out_size; (void)ws_size;
    const float* Q    = (const float*)d_in[0];
    const float* K    = (const float*)d_in[1];
    const float* V    = (const float*)d_in[2];
    const unsigned char* mask = (const unsigned char*)d_in[3];   // jax bool -> 1 byte
    const float* W_Q  = (const float*)d_in[4];
    const float* W_K  = (const float*)d_in[5];
    const float* W_V  = (const float*)d_in[6];
    const float* W_fc = (const float*)d_in[7];

    float* pred = (float*)d_out;                          // [B,S,EMB]
    float* attn = (float*)d_out + (size_t)MTOT * EMB;     // [B,NH,S,S]

    char* ws = (char*)d_ws;
    const size_t MB = 1024 * 1024;
    _Float16* Qh    = (_Float16*)(ws);              // 16 MB each
    _Float16* Kh    = (_Float16*)(ws + 16 * MB);
    _Float16* Vh    = (_Float16*)(ws + 32 * MB);
    _Float16* WQt   = (_Float16*)(ws + 48 * MB);    // 2 MB each (transposed f16)
    _Float16* WKt   = (_Float16*)(ws + 50 * MB);
    _Float16* WVt   = (_Float16*)(ws + 52 * MB);
    _Float16* WFt   = (_Float16*)(ws + 54 * MB);
    _Float16* qbuf  = (_Float16*)(ws + 56 * MB);    // [B*H][S][64] f16
    _Float16* kbuf  = (_Float16*)(ws + 72 * MB);    // [B*H][S][64] f16
    _Float16* vtb   = (_Float16*)(ws + 88 * MB);    // [B*H][64][S] f16 (transposed)
    _Float16* predh = (_Float16*)(ws + 104 * MB);   // [B*S][1024] f16

    const int nElem = MTOT * EMB;                   // 8388608
    cvt_f32_f16<<<(nElem + 255) / 256, 256, 0, stream>>>(Q, Qh, nElem);
    cvt_f32_f16<<<(nElem + 255) / 256, 256, 0, stream>>>(K, Kh, nElem);
    cvt_f32_f16<<<(nElem + 255) / 256, 256, 0, stream>>>(V, Vh, nElem);
    transpose_cvt<<<4096, 256, 0, stream>>>(W_Q,  WQt);
    transpose_cvt<<<4096, 256, 0, stream>>>(W_K,  WKt);
    transpose_cvt<<<4096, 256, 0, stream>>>(W_V,  WVt);
    transpose_cvt<<<4096, 256, 0, stream>>>(W_fc, WFt);

    dim3 gg(EMB / 64, MTOT / 128);                  // (16, 64)
    gemm_wmma<<<gg, 256, 0, stream>>>(Qh, WQt, qbuf, EMB, 1);
    gemm_wmma<<<gg, 256, 0, stream>>>(Kh, WKt, kbuf, EMB, 1);
    gemm_wmma<<<gg, 256, 0, stream>>>(Vh, WVt, vtb,  EMB, 2);

    dim3 ga(SEQ / 16, BAT * NH);                    // (128, 64)
    attn_fused<<<ga, 256, SMEM_BYTES, stream>>>(qbuf, kbuf, vtb, mask, attn, predh);

    gemm_wmma<<<gg, 256, 0, stream>>>(predh, WFt, pred, EMB, 0);
}
